// LSTMAttnDecoder_23141283791681
// MI455X (gfx1250) — matrix operand order repaired and checked by reading, not compiled, not run on hardware
//
#include <hip/hip_runtime.h>
#include <hip/hip_bf16.h>

// ---------------------------------------------------------------------------
// LSTM + Bahdanau-attention decoder for MI455X (gfx1250, wave32, WMMA).
// All GEMMs run as bf16 x bf16 -> fp32 via v_wmma_f32_16x16x32_bf16 with
// compile-time K (full unroll -> compiler-scheduled load/WMMA overlap, no
// rotation copies) and dual accumulator chains.
// Per-step GEMMs use split-K (4 slices) + combine to widen per-step
// parallelism, since the T=128 recurrence serializes steps.
// ---------------------------------------------------------------------------

#define VSZ   10000
#define D_IMG 768
#define D_H   512
#define BB    16
#define NN    196
#define TT    128
#define KSL   4        // split-K slices for per-step GEMMs

typedef __attribute__((ext_vector_type(16))) __bf16 v16bf;
typedef __attribute__((ext_vector_type(8)))  float  v8f;

union frag { unsigned u[8]; v16bf v; };

static __device__ __forceinline__ unsigned short f32_to_bf16_rne(float f) {
    unsigned u = __float_as_uint(f);
    u += 0x7FFFu + ((u >> 16) & 1u);     // round-to-nearest-even
    return (unsigned short)(u >> 16);
}

// Fragment loads (16-bit A 16x32 / B 32x16), contiguous k = 16h + s for A and B alike.
// A: lane = M row, K pairs at h*16 + 2r
// B (fed as rows of W = N x K): lane = N col, K pairs at h*16 + 2r
static __device__ __forceinline__ void load_frags(frag& fa, frag& fb,
                                                  const unsigned short* __restrict__ Arow,
                                                  const unsigned short* __restrict__ Wrow,
                                                  int kk, int h) {
#pragma unroll
    for (int r = 0; r < 8; ++r) {
        const int ka = kk + (h << 4) + (r << 1);
        fa.u[r] = *(const unsigned*)(Arow + ka);
        const int kb = kk + (h << 4) + (r << 1);
        fb.u[r] = *(const unsigned*)(Wrow + kb);
    }
}

// ---------------------------------------------------------------------------
// Fully-unrolled K-range MAC: acc over A[row, k0:k0+KLEN] x W[col, ...]^T.
// KLEN is compile-time and a multiple of 64; the unrolled chunks each get
// their own registers so the scheduler overlaps b128 loads with WMMAs
// without rotation copies. Two independent accumulator chains.
// ---------------------------------------------------------------------------
template <int KLEN>
static __device__ __forceinline__ v8f gemm_krange(const unsigned short* __restrict__ Arow,
                                                  const unsigned short* __restrict__ Wrow,
                                                  int k0, int h) {
    static_assert(KLEN % 64 == 0, "KLEN must be a multiple of 64");
    v8f acc0 = {}, acc1 = {};
#pragma unroll
    for (int kk = 0; kk < KLEN; kk += 64) {
        frag a0, b0, a1, b1;
        load_frags(a0, b0, Arow, Wrow, k0 + kk, h);
        load_frags(a1, b1, Arow, Wrow, k0 + kk + 32, h);
        acc0 = __builtin_amdgcn_wmma_f32_16x16x32_bf16(
            false, a0.v, false, b0.v, (short)0, acc0, false, false);
        acc1 = __builtin_amdgcn_wmma_f32_16x16x32_bf16(
            false, a1.v, false, b1.v, (short)0, acc1, false, false);
    }
    return acc0 + acc1;
}

// ---------------------------------------------------------------------------
// fp32 -> bf16 (stored as u16) conversion, one element per thread
// ---------------------------------------------------------------------------
__global__ void cvt_bf16_kernel(const float* __restrict__ src,
                                unsigned short* __restrict__ dst, int n) {
    int i = blockIdx.x * blockDim.x + threadIdx.x;
    if (i < n) dst[i] = f32_to_bf16_rne(src[i]);
}

// ---------------------------------------------------------------------------
// Wcat[n, 0:1024] = Wih[n,:], Wcat[n, 1024:1536] = Whh[n,:]  (bf16)
// ---------------------------------------------------------------------------
__global__ void build_wcat_kernel(const float* __restrict__ Wih,
                                  const float* __restrict__ Whh,
                                  unsigned short* __restrict__ Wcat) {
    int idx = blockIdx.x * blockDim.x + threadIdx.x;
    if (idx >= 4 * D_H * 3 * D_H) return;
    int n = idx / (3 * D_H), k = idx % (3 * D_H);
    float v = (k < 2 * D_H) ? Wih[(size_t)n * 2 * D_H + k]
                            : Whh[(size_t)n * D_H + (k - 2 * D_H)];
    Wcat[idx] = f32_to_bf16_rne(v);
}

// ---------------------------------------------------------------------------
// Direct WMMA GEMM:  C[M,N] = A[M,K](bf16) * W[N,K](bf16)^T + bias0
// One wave per 16x16 C tile. Used for the large one-time precompute GEMMs.
// ---------------------------------------------------------------------------
template <int K>
__global__ void gemm_bf16_kernel(const unsigned short* __restrict__ A,
                                 const unsigned short* __restrict__ W,
                                 float* __restrict__ C,
                                 const float* __restrict__ bias0,
                                 int M, int N, long long ldC) {
    const int wave   = (int)((blockIdx.x * blockDim.x + threadIdx.x) >> 5);
    const int lane   = threadIdx.x & 31;
    const int tilesN = N >> 4;
    const int tilesM = M >> 4;
    if (wave >= tilesM * tilesN) return;        // wave-uniform: EXEC stays full
    const int tm = wave / tilesN;
    const int tn = wave % tilesN;

    const int m = lane & 15;
    const int h = lane >> 4;

    const unsigned short* Arow = A + (size_t)(tm * 16 + m) * K;
    const unsigned short* Wrow = W + (size_t)(tn * 16 + m) * K;

    v8f acc = gemm_krange<K>(Arow, Wrow, 0, h);

    const int col = tn * 16 + (lane & 15);
    const float bias = bias0 ? bias0[col] : 0.0f;
#pragma unroll
    for (int r = 0; r < 8; ++r) {
        const int row = tm * 16 + (h << 3) + r;
        C[(size_t)row * (size_t)ldC + col] = acc[r] + bias;
    }
}

// ---------------------------------------------------------------------------
// Split-K WMMA GEMM: partial[s][M][N] = A[:, s*K/ks : (s+1)*K/ks] * W^T slice.
// One wave per (16x16 tile, K-slice). K/KSLICES must be a multiple of 64.
// ---------------------------------------------------------------------------
template <int K, int KSLICES>
__global__ void gemm_bf16_splitk_kernel(const unsigned short* __restrict__ A,
                                        const unsigned short* __restrict__ W,
                                        float* __restrict__ partial,
                                        int M, int N) {
    const int wave   = (int)((blockIdx.x * blockDim.x + threadIdx.x) >> 5);
    const int lane   = threadIdx.x & 31;
    const int tilesN = N >> 4;
    const int tilesM = M >> 4;
    if (wave >= tilesM * tilesN * KSLICES) return;   // wave-uniform
    const int tile = wave / KSLICES;
    const int s    = wave % KSLICES;
    const int tm = tile / tilesN;
    const int tn = tile % tilesN;
    constexpr int kLen = K / KSLICES;
    const int k0 = s * kLen;

    const int m = lane & 15;
    const int h = lane >> 4;

    const unsigned short* Arow = A + (size_t)(tm * 16 + m) * K;
    const unsigned short* Wrow = W + (size_t)(tn * 16 + m) * K;

    v8f acc = gemm_krange<kLen>(Arow, Wrow, k0, h);

    const int col = tn * 16 + (lane & 15);
#pragma unroll
    for (int r = 0; r < 8; ++r) {
        const int row = tm * 16 + (h << 3) + r;
        partial[((size_t)s * M + row) * N + col] = acc[r];
    }
}

// ---------------------------------------------------------------------------
// combine: C[row, col] = sum_s partial[s][row][col] + bias0 + bias1
// ---------------------------------------------------------------------------
__global__ void combine_kernel(const float* __restrict__ partial,
                               float* __restrict__ C,
                               const float* __restrict__ bias0,
                               const float* __restrict__ bias1,
                               int M, int N, long long ldC, int kslices) {
    int idx = blockIdx.x * blockDim.x + threadIdx.x;
    if (idx >= M * N) return;
    int row = idx / N, col = idx % N;
    float v = 0.0f;
    for (int s = 0; s < kslices; ++s)
        v += partial[((size_t)s * M + row) * N + col];
    if (bias0) v += bias0[col];
    if (bias1) v += bias1[col];
    C[(size_t)row * (size_t)ldC + col] = v;
}

// ---------------------------------------------------------------------------
// h0 = cls @ ih_w^T + ih_b ; c0 = cls @ ic_w^T + ic_b   (tiny, fp32 VALU)
// ---------------------------------------------------------------------------
__global__ void init_hc_kernel(const float* __restrict__ cls,
                               const float* __restrict__ ih_w,
                               const float* __restrict__ ih_b,
                               const float* __restrict__ ic_w,
                               const float* __restrict__ ic_b,
                               float* __restrict__ h, float* __restrict__ c) {
    int idx = blockIdx.x * blockDim.x + threadIdx.x;
    if (idx >= BB * D_H) return;
    int b = idx >> 9, j = idx & (D_H - 1);
    const float* x  = cls + (size_t)b * D_IMG;
    const float* wh = ih_w + (size_t)j * D_IMG;
    const float* wc = ic_w + (size_t)j * D_IMG;
    float ah = ih_b[j], ac = ic_b[j];
    for (int k = 0; k < D_IMG; ++k) {
        float xv = x[k];
        ah = fmaf(xv, wh[k], ah);
        ac = fmaf(xv, wc[k], ac);
    }
    h[idx] = ah;
    c[idx] = ac;
}

// ---------------------------------------------------------------------------
// scores[b,n] = sum_d v[d] * tanh(hWh[b,d] + keysWk[b,n,d])   (wave per (b,n))
// ---------------------------------------------------------------------------
__global__ void attn_scores_kernel(const float* __restrict__ hWh,
                                   const float* __restrict__ keysWk,
                                   const float* __restrict__ vvec,
                                   float* __restrict__ scores) {
    int gw   = (int)((blockIdx.x * blockDim.x + threadIdx.x) >> 5);
    int lane = threadIdx.x & 31;
    if (gw >= BB * NN) return;
    int b = gw / NN;
    const float* hw = hWh + (size_t)b * D_H;
    const float* kw = keysWk + (size_t)gw * D_H;
    float acc = 0.0f;
    for (int k = lane; k < D_H; k += 32)
        acc = fmaf(vvec[k], tanhf(hw[k] + kw[k]), acc);
#pragma unroll
    for (int off = 16; off; off >>= 1)
        acc += __shfl_down(acc, off, 32);
    if (lane == 0) scores[gw] = acc;
}

// ---------------------------------------------------------------------------
// Per batch row: softmax over N=196 then ctx[b,:] = alpha @ keys[b]  (block/b)
// ---------------------------------------------------------------------------
__global__ void softmax_ctx_kernel(const float* __restrict__ scores,
                                   const float* __restrict__ keys,
                                   float* __restrict__ ctx) {
    __shared__ float alpha[NN];
    __shared__ float red[256];
    const int b = blockIdx.x, tid = threadIdx.x;

    float s = (tid < NN) ? scores[b * NN + tid] : -3.4e38f;
    red[tid] = s; __syncthreads();
    for (int off = 128; off; off >>= 1) {
        if (tid < off) red[tid] = fmaxf(red[tid], red[tid + off]);
        __syncthreads();
    }
    const float mx = red[0]; __syncthreads();

    float e = (tid < NN) ? __expf(s - mx) : 0.0f;
    red[tid] = e; __syncthreads();
    for (int off = 128; off; off >>= 1) {
        if (tid < off) red[tid] += red[tid + off];
        __syncthreads();
    }
    const float inv = 1.0f / red[0];
    if (tid < NN) alpha[tid] = e * inv;
    __syncthreads();

    for (int d = tid; d < D_H; d += 256) {
        float a = 0.0f;
        const float* kb = keys + ((size_t)b * NN) * D_H + d;
        for (int n = 0; n < NN; ++n)
            a = fmaf(alpha[n], kb[(size_t)n * D_H], a);
        ctx[b * D_H + d] = a;
    }
}

// ---------------------------------------------------------------------------
// zh[b] = bf16(concat(emb[tgt_ids[b,t]], ctx[b], h[b]))   (16 x 1536)
// ---------------------------------------------------------------------------
__global__ void build_zh_kernel(const int* __restrict__ tgt_ids,
                                const float* __restrict__ emb,
                                const float* __restrict__ ctx,
                                const unsigned short* __restrict__ h_bf,
                                unsigned short* __restrict__ zh_bf, int t) {
    int idx = blockIdx.x * blockDim.x + threadIdx.x;
    if (idx >= BB * 3 * D_H) return;
    int b = idx / (3 * D_H), j = idx % (3 * D_H);
    unsigned short r;
    if (j < D_H)
        r = f32_to_bf16_rne(emb[(size_t)tgt_ids[b * TT + t] * D_H + j]);
    else if (j < 2 * D_H)
        r = f32_to_bf16_rne(ctx[b * D_H + (j - D_H)]);
    else
        r = h_bf[b * D_H + (j - 2 * D_H)];
    zh_bf[idx] = r;
}

// ---------------------------------------------------------------------------
// LSTM cell + LayerNorm. gates layout [i|f|g|o] along 4*D_H. Block per b.
// ---------------------------------------------------------------------------
__global__ void lstm_ln_kernel(const float* __restrict__ gates,
                               float* __restrict__ c,
                               float* __restrict__ h,
                               unsigned short* __restrict__ h_bf,
                               const float* __restrict__ ln_g,
                               const float* __restrict__ ln_b) {
    __shared__ float hraw[D_H];
    __shared__ float red[256];
    const int b = blockIdx.x, tid = threadIdx.x;
    const float* gb = gates + (size_t)b * 4 * D_H;

    for (int d = tid; d < D_H; d += 256) {
        float gi = gb[d];
        float gf = gb[D_H + d];
        float gg = gb[2 * D_H + d];
        float go = gb[3 * D_H + d];
        float si = 1.0f / (1.0f + __expf(-gi));
        float sf = 1.0f / (1.0f + __expf(-gf));
        float so = 1.0f / (1.0f + __expf(-go));
        float cn = sf * c[b * D_H + d] + si * tanhf(gg);
        c[b * D_H + d] = cn;
        hraw[d] = so * tanhf(cn);
    }
    __syncthreads();

    red[tid] = hraw[tid] + hraw[tid + 256]; __syncthreads();
    for (int off = 128; off; off >>= 1) {
        if (tid < off) red[tid] += red[tid + off];
        __syncthreads();
    }
    const float mu = red[0] * (1.0f / D_H); __syncthreads();

    float d0 = hraw[tid] - mu, d1 = hraw[tid + 256] - mu;
    red[tid] = d0 * d0 + d1 * d1; __syncthreads();
    for (int off = 128; off; off >>= 1) {
        if (tid < off) red[tid] += red[tid + off];
        __syncthreads();
    }
    const float rstd = rsqrtf(red[0] * (1.0f / D_H) + 1e-5f);

    for (int d = tid; d < D_H; d += 256) {
        float hn = (hraw[d] - mu) * rstd * ln_g[d] + ln_b[d];
        h[b * D_H + d] = hn;
        h_bf[b * D_H + d] = f32_to_bf16_rne(hn);
    }
}

// ---------------------------------------------------------------------------
// Host launcher
// ---------------------------------------------------------------------------
extern "C" void kernel_launch(void* const* d_in, const int* in_sizes, int n_in,
                              void* d_out, int out_size, void* d_ws, size_t ws_size,
                              hipStream_t stream) {
    (void)in_sizes; (void)n_in; (void)out_size; (void)ws_size;

    const float* patches = (const float*)d_in[0];
    const float* cls     = (const float*)d_in[1];
    const int*   tgt_ids = (const int*)  d_in[2];
    const float* kv_w    = (const float*)d_in[3];
    const float* kv_b    = (const float*)d_in[4];
    const float* ih_w    = (const float*)d_in[5];
    const float* ih_b    = (const float*)d_in[6];
    const float* ic_w    = (const float*)d_in[7];
    const float* ic_b    = (const float*)d_in[8];
    const float* emb     = (const float*)d_in[9];
    const float* attn_Wh = (const float*)d_in[10];
    const float* attn_Wk = (const float*)d_in[11];
    const float* attn_v  = (const float*)d_in[12];
    const float* lstm_Wih= (const float*)d_in[13];
    const float* lstm_Whh= (const float*)d_in[14];
    const float* lstm_bih= (const float*)d_in[15];
    const float* lstm_bhh= (const float*)d_in[16];
    const float* ln_g    = (const float*)d_in[17];
    const float* ln_b    = (const float*)d_in[18];
    const float* out_w   = (const float*)d_in[19];
    const float* out_b   = (const float*)d_in[20];
    float* out = (float*)d_out;

    // ---- workspace carve-out (256B aligned) ----
    char* ws = (char*)d_ws;
    size_t off = 0;
    auto carve = [&](size_t bytes) -> char* {
        char* p = ws + off;
        off = (off + bytes + 255) & ~(size_t)255;
        return p;
    };
    const int BN = BB * NN;  // 3136
    unsigned short* patches_bf = (unsigned short*)carve((size_t)BN * D_IMG * 2);
    unsigned short* outw_bf    = (unsigned short*)carve((size_t)VSZ * D_H * 2);
    unsigned short* Wcat_bf    = (unsigned short*)carve((size_t)4 * D_H * 3 * D_H * 2);
    unsigned short* kvw_bf     = (unsigned short*)carve((size_t)D_H * D_IMG * 2);
    unsigned short* Wk_bf      = (unsigned short*)carve((size_t)D_H * D_H * 2);
    unsigned short* Wh_bf      = (unsigned short*)carve((size_t)D_H * D_H * 2);
    float*          keys       = (float*)carve((size_t)BN * D_H * 4);
    unsigned short* keys_bf    = (unsigned short*)carve((size_t)BN * D_H * 2);
    float*          keysWk     = (float*)carve((size_t)BN * D_H * 4);
    float*          h_f        = (float*)carve((size_t)BB * D_H * 4);
    float*          c_f        = (float*)carve((size_t)BB * D_H * 4);
    unsigned short* h_bf       = (unsigned short*)carve((size_t)BB * D_H * 2);
    float*          hWh        = (float*)carve((size_t)BB * D_H * 4);
    float*          scores     = (float*)carve((size_t)BB * NN * 4);
    float*          ctx        = (float*)carve((size_t)BB * D_H * 4);
    unsigned short* zh_bf      = (unsigned short*)carve((size_t)BB * 3 * D_H * 2);
    float*          gates      = (float*)carve((size_t)BB * 4 * D_H * 4);
    float*          partial    = (float*)carve((size_t)KSL * BB * VSZ * 4);

    auto cvt = [&](const float* s, unsigned short* d, int n) {
        cvt_bf16_kernel<<<(n + 255) / 256, 256, 0, stream>>>(s, d, n);
    };
    auto waveBlocks = [](int waves) { return (waves * 32 + 255) / 256; };

    // ---- one-time precompute ----
    cvt(patches, patches_bf, BN * D_IMG);
    cvt(out_w,   outw_bf,    VSZ * D_H);
    cvt(kv_w,    kvw_bf,     D_H * D_IMG);
    cvt(attn_Wk, Wk_bf,      D_H * D_H);
    cvt(attn_Wh, Wh_bf,      D_H * D_H);
    build_wcat_kernel<<<(4 * D_H * 3 * D_H + 255) / 256, 256, 0, stream>>>(
        lstm_Wih, lstm_Whh, Wcat_bf);

    init_hc_kernel<<<(BB * D_H + 255) / 256, 256, 0, stream>>>(
        cls, ih_w, ih_b, ic_w, ic_b, h_f, c_f);
    cvt(h_f, h_bf, BB * D_H);

    // keys = patches @ kv_w^T + kv_b   (3136 x 512, K=768) : 6272 waves
    gemm_bf16_kernel<D_IMG><<<waveBlocks((BN / 16) * (D_H / 16)), 256, 0, stream>>>(
        patches_bf, kvw_bf, keys, kv_b, BN, D_H, D_H);
    cvt(keys, keys_bf, BN * D_H);
    // keysWk = keys @ attn_Wk^T        (3136 x 512, K=512)
    gemm_bf16_kernel<D_H><<<waveBlocks((BN / 16) * (D_H / 16)), 256, 0, stream>>>(
        keys_bf, Wk_bf, keysWk, nullptr, BN, D_H, D_H);

    // ---- sequential decode loop ----
    for (int t = 0; t < TT; ++t) {
        // hWh = h @ attn_Wh^T  (16 x 512, K=512) : 32 tiles x 4 slices
        gemm_bf16_splitk_kernel<D_H, KSL><<<waveBlocks(32 * KSL), 256, 0, stream>>>(
            h_bf, Wh_bf, partial, BB, D_H);
        combine_kernel<<<(BB * D_H + 255) / 256, 256, 0, stream>>>(
            partial, hWh, nullptr, nullptr, BB, D_H, D_H, KSL);

        // attention scores: 3136 waves
        attn_scores_kernel<<<waveBlocks(BB * NN), 256, 0, stream>>>(
            hWh, keysWk, attn_v, scores);

        softmax_ctx_kernel<<<BB, 256, 0, stream>>>(scores, keys, ctx);

        build_zh_kernel<<<(BB * 3 * D_H + 255) / 256, 256, 0, stream>>>(
            tgt_ids, emb, ctx, h_bf, zh_bf, t);

        // gates = [x|ctx|h] @ [Wih|Whh]^T + bih + bhh
        // (16 x 2048, K=1536) : 128 tiles x 4 slices = 512 waves
        gemm_bf16_splitk_kernel<3 * D_H, KSL><<<waveBlocks(128 * KSL), 256, 0, stream>>>(
            zh_bf, Wcat_bf, partial, BB, 4 * D_H);
        combine_kernel<<<(BB * 4 * D_H + 255) / 256, 256, 0, stream>>>(
            partial, gates, lstm_bih, lstm_bhh, BB, 4 * D_H, 4 * D_H, KSL);

        lstm_ln_kernel<<<BB, 256, 0, stream>>>(gates, c_f, h_f, h_bf, ln_g, ln_b);

        // logits_t = h @ out_w^T + out_b
        // (16 x 10000, K=512) : 625 tiles x 4 slices = 2500 waves
        gemm_bf16_splitk_kernel<D_H, KSL><<<waveBlocks(625 * KSL), 256, 0, stream>>>(
            h_bf, outw_bf, partial, BB, VSZ);
        combine_kernel<<<(BB * VSZ + 255) / 256, 256, 0, stream>>>(
            partial, out + (size_t)t * VSZ, out_b, nullptr,
            BB, VSZ, (long long)TT * VSZ, KSL);
    }
}
